// RNNBase_80607946211411
// MI455X (gfx1250) — matrix operand order repaired and checked by reading, not compiled, not run on hardware
//
#include <hip/hip_runtime.h>
#include <math.h>

#define S_LEN 128
#define B_SZ  128
#define H_DIM 1024
#define V_SZ  10000
#define L_NUM 2
#define NBLK  (H_DIM / 16)    // persistent grid: 64 blocks
#define KC    128             // logits K-chunk

typedef __attribute__((ext_vector_type(16))) __bf16 v16bf;
typedef __attribute__((ext_vector_type(8)))  __bf16 v8bf;
typedef __attribute__((ext_vector_type(8)))  float  v8f;

// ---------------- CDNA5 async global->LDS path (guarded) ----------------
#if defined(__has_builtin)
#if __has_builtin(__builtin_amdgcn_global_load_async_to_lds_b128) && \
    __has_builtin(__builtin_amdgcn_s_wait_asynccnt)
#define HAVE_ASYNC_LDS 1
#endif
#endif

typedef int v4i __attribute__((vector_size(16)));
typedef __attribute__((address_space(1))) v4i gv4i;   // global (AS1) 128-bit
typedef __attribute__((address_space(3))) v4i lv4i;   // LDS    (AS3) 128-bit

__device__ __forceinline__ void copy16_to_lds(void* sdst, const void* gsrc) {
#ifdef HAVE_ASYNC_LDS
    __builtin_amdgcn_global_load_async_to_lds_b128((gv4i*)gsrc, (lv4i*)sdst, 0, 0);
#else
    *(uint4*)sdst = *(const uint4*)gsrc;
#endif
}
__device__ __forceinline__ void async_join() {
#ifdef HAVE_ASYNC_LDS
    __builtin_amdgcn_s_wait_asynccnt(0);
#endif
}

// ---------------- helpers ----------------

__device__ __forceinline__ __bf16 f2bf(float f) {
    unsigned u = __float_as_uint(f);
    unsigned r = (u + 0x7FFFu + ((u >> 16) & 1u)) >> 16;   // RNE
    unsigned short s = (unsigned short)r;
    return __builtin_bit_cast(__bf16, s);
}

// A-fragment (16x32 bf16 MxK): lane l holds row M0+(l&15); halves 0..7 at
// k0+(l>>4)*8, halves 8..15 at k0+16+(l>>4)*8   (ISA 7.12.2)
__device__ __forceinline__ v16bf load_a16(const __bf16* __restrict__ p) {
    v8bf lo = *(const v8bf*)p;
    v8bf hi = *(const v8bf*)(p + 16);
    v16bf a;
#pragma unroll
    for (int i = 0; i < 8; ++i) { a[i] = lo[i]; a[i + 8] = hi[i]; }
    return a;
}

__device__ __forceinline__ v8f wmma_bf16(v16bf a, v16bf b, v8f c) {
    return __builtin_amdgcn_wmma_f32_16x16x32_bf16(false, a, false, b, (short)0, c, false, false);
}

// device-wide barrier for the persistent RNN kernel (all NBLK blocks resident)
__device__ __forceinline__ void grid_sync(unsigned* bar, unsigned nblocks, unsigned phase) {
    __syncthreads();
    if (threadIdx.x == 0) {
        __threadfence();  // release our global writes to agent scope
        __hip_atomic_fetch_add(bar, 1u, __ATOMIC_RELEASE, __HIP_MEMORY_SCOPE_AGENT);
        while (__hip_atomic_load(bar, __ATOMIC_ACQUIRE, __HIP_MEMORY_SCOPE_AGENT) < nblocks * phase)
            __builtin_amdgcn_s_sleep(2);
    }
    __syncthreads();
    __threadfence();      // acquire: refresh near caches before reading peers' data
}

// ---------------- prep kernels ----------------

__global__ void k_zero_u32(unsigned* p) { *p = 0u; }

// out[b][c][r] = bf16(in[b][r][c])  -> K-major bf16 weights
__global__ void k_transpose_bf16(const float* __restrict__ in, __bf16* __restrict__ out,
                                 int rows, int cols, int nb) {
    size_t i = (size_t)blockIdx.x * blockDim.x + threadIdx.x;
    size_t per = (size_t)rows * cols;
    if (i >= per * (size_t)nb) return;
    size_t b = i / per, rem = i % per;
    size_t c = rem / rows, r = rem % rows;
    out[i] = f2bf(in[b * per + r * (size_t)cols + c]);
}

__global__ void k_f32_to_bf16(const float* __restrict__ in, __bf16* __restrict__ out, size_t n) {
    size_t i = (size_t)blockIdx.x * blockDim.x + threadIdx.x;
    if (i < n) out[i] = f2bf(in[i]);
}

__global__ void k_embed(const int* __restrict__ tok, const float* __restrict__ emb,
                        __bf16* __restrict__ x) {
    size_t i = (size_t)blockIdx.x * blockDim.x + threadIdx.x;   // over S*B*H
    int h = (int)(i & (H_DIM - 1));
    size_t sb = i >> 10;
    int id = tok[sb];
    x[i] = f2bf(emb[(size_t)id * H_DIM + h]);
}

// ---------------- persistent recurrence ----------------
// 64 blocks x 256 threads; block owns 16 N-columns, wave owns 16 M-rows.
// Loops all (t, layer) phases internally; device barrier between phases.
// Hidden state double buffer is ONE allocation hb[2][L][B][H] indexed by
// parity offset so every state access derives from a single pointer base
// (keeps loads on the global path, not flat).
__global__ __launch_bounds__(256) void k_rnn_persistent(
    const __bf16* __restrict__ xb,     // [S,B,H]
    const __bf16* __restrict__ wxt,    // [L,H,H] K-major
    const __bf16* __restrict__ wht,    // [L,H,H] K-major
    const float*  __restrict__ bh,     // [L,H]
    __bf16* __restrict__ hb,           // [2,L,B,H]; parity 0 holds bf16(hidden) at entry
    __bf16* __restrict__ topb,         // [S,B,H]
    float*  __restrict__ fstate,       // [L,B,H] (tail of d_out)
    unsigned* __restrict__ bar)
{
    __shared__ __align__(32) __bf16 sWx[H_DIM * 16];
    __shared__ __align__(32) __bf16 sWh[H_DIM * 16];
    const int Nb   = blockIdx.x * 16;
    const int wave = threadIdx.x >> 5;
    const int lane = threadIdx.x & 31;
    const int M0   = wave * 16;
    const int kh   = (lane >> 4) & 1;
    const int rowA = M0 + (lane & 15);
    const int n    = Nb + (lane & 15);
    const int mrow = M0 + ((lane >> 4) << 3);
    const size_t BH  = (size_t)B_SZ * H_DIM;
    const size_t HH  = (size_t)H_DIM * H_DIM;
    const size_t LBH = (size_t)L_NUM * BH;

    unsigned phase = 0;
    for (int t = 0; t < S_LEN; ++t) {
        const size_t oldOff = (size_t)(t & 1) * LBH;
        const size_t newOff = (size_t)((t + 1) & 1) * LBH;
#pragma unroll
        for (int l = 0; l < L_NUM; ++l) {
            const __bf16* X  = (l == 0) ? (xb + (size_t)t * BH)   // layer0: embedded token
                                        : (hb + newOff);          // layer1: new h[0]
            const __bf16* Hp = hb + oldOff + (size_t)l * BH;
            const __bf16* wx = wxt + (size_t)l * HH;
            const __bf16* wh = wht + (size_t)l * HH;

            // stage this layer's K=1024 x N=16 panels into LDS (async DMA)
            for (int r = threadIdx.x; r < H_DIM; r += 256) {
                copy16_to_lds(sWx + r * 16,     wx + (size_t)r * H_DIM + Nb);
                copy16_to_lds(sWx + r * 16 + 8, wx + (size_t)r * H_DIM + Nb + 8);
                copy16_to_lds(sWh + r * 16,     wh + (size_t)r * H_DIM + Nb);
                copy16_to_lds(sWh + r * 16 + 8, wh + (size_t)r * H_DIM + Nb + 8);
            }
            async_join();
            __syncthreads();

            const __bf16* pX = X  + (size_t)rowA * H_DIM + kh * 8;
            const __bf16* pH = Hp + (size_t)rowA * H_DIM + kh * 8;
            v8f acc = {};
#pragma unroll 4
            for (int k0 = 0; k0 < H_DIM; k0 += 32) {
                v16bf a = load_a16(pX + k0);
                v16bf b;
#pragma unroll
                for (int i = 0; i < 16; ++i) b[i] = sWx[(size_t)(k0 + kh * 8 + (i & 7) + (i >> 3) * 16) * 16 + (lane & 15)];
                acc = wmma_bf16(a, b, acc);
                a = load_a16(pH + k0);
#pragma unroll
                for (int i = 0; i < 16; ++i) b[i] = sWh[(size_t)(k0 + kh * 8 + (i & 7) + (i >> 3) * 16) * 16 + (lane & 15)];
                acc = wmma_bf16(a, b, acc);
            }

            const float bv = bh[l * H_DIM + n];
            __bf16* Hn = hb + newOff + (size_t)l * BH;
#pragma unroll
            for (int j = 0; j < 8; ++j) {
                float v = tanhf(acc[j] + bv);
                __bf16 hbv = f2bf(v);
                size_t idx = (size_t)(mrow + j) * H_DIM + n;
                Hn[idx] = hbv;
                if (l == L_NUM - 1)
                    __builtin_nontemporal_store(hbv, topb + (size_t)t * BH + idx);
                if (t == S_LEN - 1)
                    __builtin_nontemporal_store(v, fstate + (size_t)l * BH + idx);
            }
            ++phase;
            grid_sync(bar, NBLK, phase);
        }
    }
}

// ---------------- logits: out[SB,V] = top[SB,H] * WoutT[H,V] + bout ----------------
// grid (125, 128): block = 128 M-rows x 80 N-cols; wave = 16 M x 5 N-tiles.
// Weight panel double-buffered in LDS by K-chunks of 128 (async staged).
__global__ __launch_bounds__(256) void k_logits(
    const __bf16* __restrict__ A,     // [S*B, H]
    const __bf16* __restrict__ WoT,   // [H, V] K-major
    const float*  __restrict__ bout,  // [V]
    float* __restrict__ out)          // [S*B, V]
{
    __shared__ __align__(32) __bf16 sW[2][KC * 80];   // 2 x 20 KB
    const int Nb   = blockIdx.x * 80;
    const int wave = threadIdx.x >> 5;
    const int lane = threadIdx.x & 31;
    const size_t M0 = ((size_t)blockIdx.y * 8 + wave) * 16;
    const int kh = (lane >> 4) & 1;
    const __bf16* pA = A + (M0 + (lane & 15)) * H_DIM + kh * 8;

    const int sr = threadIdx.x >> 1;          // row 0..127 within chunk
    const int sh = (threadIdx.x & 1) * 40;    // 80-byte half of the 160B row
    auto stage = [&](int buf, int c) {
        const __bf16* g = WoT + (size_t)(c * KC + sr) * V_SZ + Nb + sh;
        __bf16* s = &sW[buf][sr * 80 + sh];
#pragma unroll
        for (int q = 0; q < 5; ++q)
            copy16_to_lds(s + q * 8, g + q * 8);
    };

    v8f acc[5] = {};
    stage(0, 0);
    for (int c = 0; c < H_DIM / KC; ++c) {
        async_join();
        __syncthreads();                       // chunk c staged, prev reads done
        if (c + 1 < H_DIM / KC) stage((c + 1) & 1, c + 1);
        const __bf16* sb = sW[c & 1];
#pragma unroll 2
        for (int kk = 0; kk < KC; kk += 32) {
            v16bf a = load_a16(pA + c * KC + kk);
#pragma unroll
            for (int nt = 0; nt < 5; ++nt) {
                v16bf b;
#pragma unroll
                for (int i = 0; i < 16; ++i) b[i] = sb[(size_t)(kk + kh * 8 + (i & 7) + (i >> 3) * 16) * 80 + nt * 16 + (lane & 15)];
                acc[nt] = wmma_bf16(a, b, acc[nt]);
            }
        }
        __syncthreads();
    }

    const size_t mrow = M0 + ((lane >> 4) << 3);
#pragma unroll
    for (int nt = 0; nt < 5; ++nt) {
        int nn = Nb + nt * 16 + (lane & 15);
        float bv = bout[nn];
#pragma unroll
        for (int j = 0; j < 8; ++j)     // 655 MB stream: NT stores, keep L2 for reuse
            __builtin_nontemporal_store(acc[nt][j] + bv, out + (mrow + j) * V_SZ + nn);
    }
}

// ---------------- host ----------------

extern "C" void kernel_launch(void* const* d_in, const int* in_sizes, int n_in,
                              void* d_out, int out_size, void* d_ws, size_t ws_size,
                              hipStream_t stream) {
    const int*   tok    = (const int*)d_in[0];
    const float* hidden = (const float*)d_in[1];
    const float* emb    = (const float*)d_in[2];
    const float* Wx     = (const float*)d_in[3];
    const float* Wh     = (const float*)d_in[4];
    const float* bh     = (const float*)d_in[5];
    const float* Wout   = (const float*)d_in[6];
    const float* bout   = (const float*)d_in[7];
    float* out    = (float*)d_out;
    float* fstate = out + (size_t)S_LEN * B_SZ * V_SZ;

    char* ws = (char*)d_ws;
    size_t off = 0;
    auto alloc = [&](size_t bytes) -> char* {
        char* p = ws + off;
        off += (bytes + 255) & ~(size_t)255;
        return p;
    };
    __bf16* xb   = (__bf16*)alloc((size_t)S_LEN * B_SZ * H_DIM * 2);
    __bf16* topb = (__bf16*)alloc((size_t)S_LEN * B_SZ * H_DIM * 2);
    __bf16* wxt  = (__bf16*)alloc((size_t)L_NUM * H_DIM * H_DIM * 2);
    __bf16* wht  = (__bf16*)alloc((size_t)L_NUM * H_DIM * H_DIM * 2);
    __bf16* wot  = (__bf16*)alloc((size_t)H_DIM * V_SZ * 2);
    __bf16* hb   = (__bf16*)alloc((size_t)2 * L_NUM * B_SZ * H_DIM * 2);  // [2][L][B][H]
    unsigned* bar = (unsigned*)alloc(256);

    k_zero_u32<<<1, 1, 0, stream>>>(bar);
    {
        size_t tot = (size_t)L_NUM * H_DIM * H_DIM;
        k_transpose_bf16<<<dim3((unsigned)((tot + 255) / 256)), 256, 0, stream>>>(Wx, wxt, H_DIM, H_DIM, L_NUM);
        k_transpose_bf16<<<dim3((unsigned)((tot + 255) / 256)), 256, 0, stream>>>(Wh, wht, H_DIM, H_DIM, L_NUM);
    }
    {
        size_t tot = (size_t)H_DIM * V_SZ;
        k_transpose_bf16<<<dim3((unsigned)((tot + 255) / 256)), 256, 0, stream>>>(Wout, wot, V_SZ, H_DIM, 1);
    }
    {
        size_t n = (size_t)L_NUM * B_SZ * H_DIM;  // parity-0 state <- bf16(hidden)
        k_f32_to_bf16<<<dim3((unsigned)((n + 255) / 256)), 256, 0, stream>>>(hidden, hb, n);
    }
    {
        size_t n = (size_t)S_LEN * B_SZ * H_DIM;
        k_embed<<<dim3((unsigned)(n / 256)), 256, 0, stream>>>(tok, emb, xb);
    }

    // whole recurrence in ONE launch (64 co-resident blocks, device barrier)
    k_rnn_persistent<<<dim3(NBLK), 256, 0, stream>>>(
        xb, wxt, wht, bh, hb, topb, fstate, bar);

    // dominant output projection
    k_logits<<<dim3(V_SZ / 80, (S_LEN * B_SZ) / (16 * 8)), 256, 0, stream>>>(topb, wot, bout, out);
}
